// Encoder_73830487818453
// MI455X (gfx1250) — compile-verified, matrix operand order checked by reading, not run on hardware
//
#include <hip/hip_runtime.h>
#include <hip/hip_bf16.h>

typedef __bf16 bf16_t;
typedef __attribute__((ext_vector_type(16))) bf16_t v16bf;
typedef __attribute__((ext_vector_type(8)))  bf16_t v8bf;
typedef __attribute__((ext_vector_type(8)))  float  v8f;

#define N_NODES 50000
#define N_EDGES 1600000
#define FDIM    128
#define OUT_DIM 64
#define N_USERS 10000

// ---------------- degree / normalization ----------------
__global__ void degrees_kernel(const int* __restrict__ src, const int* __restrict__ dst,
                               float* __restrict__ degout, float* __restrict__ degin, int E) {
    int e = blockIdx.x * blockDim.x + threadIdx.x;
    if (e < E) {
        atomicAdd(&degout[src[e]], 1.0f);
        atomicAdd(&degin[dst[e]], 1.0f);
    }
}

__global__ void norm_kernel(const float* __restrict__ degout, const float* __restrict__ degin,
                            float* __restrict__ cs, float* __restrict__ cd, int N) {
    int i = blockIdx.x * blockDim.x + threadIdx.x;
    if (i < N) {
        cs[i] = rsqrtf(fmaxf(degout[i], 1.0f));
        cd[i] = rsqrtf(fmaxf(degin[i], 1.0f));
    }
}

// ---------------- pack f32 weights (K x N, row-major) into WMMA B-fragment order ----------------
// Fragment element e of lane l in tile (kt,nt):
//   K = kt*32 + (l/16)*8 + (e&7) + ((e&8)?16:0),  Ncol = nt*16 + (l&15)
// (mirrors the ISA 16-bit A-matrix layout, transposed: lane = column)
__global__ void pack_b_kernel(const float* __restrict__ W, bf16_t* __restrict__ Bp,
                              int Ktiles, int Ntiles, int N) {
    int idx = blockIdx.x * blockDim.x + threadIdx.x;
    int total = Ktiles * Ntiles * 512;
    if (idx >= total) return;
    int e  = idx & 15;
    int l  = (idx >> 4) & 31;
    int t  = idx >> 9;               // kt*Ntiles + nt
    int nt = t % Ntiles;
    int kt = t / Ntiles;
    int k  = kt * 32 + ((l >> 4) * 8) + (e & 7) + ((e & 8) ? 16 : 0);
    int n  = nt * 16 + (l & 15);
    Bp[idx] = (bf16_t)W[k * N + n];
}

// ---------------- WMMA GEMM ----------------
// out[m,n] = ACT( (sum_k A[m,k]*W[k,n]) * (SCALE? scale[m]:1) + (BIAS? bias[n]:0) )
// grid.x = M/16, blockDim.x = 32*NT (one wave per 16-wide N tile).
// A tile (16 x KT*32, f32) is converted to bf16 once per block into LDS in
// fragment order; all NT waves read their v16bf frags via ds loads.
template <int KT, int NT, bool SCALE, bool BIAS, int ACT, bool IDX>
__global__ void gemm_wmma_bf16(const float* __restrict__ A, int lda,
                               const int* __restrict__ rowidx,
                               const bf16_t* __restrict__ Bp,
                               const float* __restrict__ scale,
                               const float* __restrict__ bias,
                               float* __restrict__ out, int ldo) {
    __shared__ bf16_t aFrag[KT * 32 * 16];   // <= 4 KB of the WGP's 320 KB LDS

    const int lane  = threadIdx.x & 31;
    const int w     = threadIdx.x >> 5;      // n-tile handled by this wave
    const int mtile = blockIdx.x;

    // ---- stage A tile into LDS, fragment-ordered (8 K-values per chunk) ----
    constexpr int CPR   = KT * 4;            // 8-wide K chunks per row
    constexpr int WORK  = 16 * CPR;          // total chunks (exact multiple of blockDim)
    for (int i = threadIdx.x; i < WORK; i += NT * 32) {
        const int r = i / CPR;               // row within tile (constexpr divide)
        const int c = i % CPR;               // K chunk index
        int arow = mtile * 16 + r;
        if (IDX) arow = rowidx[arow];
        const float* p = A + (long)arow * lda + c * 8;
        const float4 x0 = *(const float4*)(p);
        const float4 x1 = *(const float4*)(p + 4);
        v8bf h;
        h[0] = (bf16_t)x0.x; h[1] = (bf16_t)x0.y; h[2] = (bf16_t)x0.z; h[3] = (bf16_t)x0.w;
        h[4] = (bf16_t)x1.x; h[5] = (bf16_t)x1.y; h[6] = (bf16_t)x1.z; h[7] = (bf16_t)x1.w;
        // ISA 16-bit A layout: lane = r (+16 for odd chunk), element half by chunk bit 1
        const int lane_s = r + ((c & 1) ? 16 : 0);
        const int ebase  = (c & 2) ? 8 : 0;
        const int kt     = c >> 2;
        *(v8bf*)&aFrag[(kt * 32 + lane_s) * 16 + ebase] = h;  // 16B-aligned ds store
    }
    __syncthreads();

    // ---- K loop: fully unrolled back-to-back WMMA ----
    v8f acc = {};
#pragma unroll
    for (int kt = 0; kt < KT; ++kt) {
        const v16bf av = *(const v16bf*)&aFrag[(kt * 32 + lane) * 16];
        const v16bf bv = *(const v16bf*)(Bp + ((long)(kt * NT + w) * 512 + lane * 16));
        acc = __builtin_amdgcn_wmma_f32_16x16x32_bf16(
            /*neg_a=*/false, av, /*neg_b=*/false, bv,
            /*c_mod=*/(short)0, acc, /*reuse_a=*/false, /*reuse_b=*/false);
    }

    // ---- epilogue: C/D layout row = r + 8*(lane/16), col = lane%16 ----
    const int col   = w * 16 + (lane & 15);
    const int mbase = mtile * 16 + (lane >> 4) * 8;
#pragma unroll
    for (int r = 0; r < 8; ++r) {
        const int grow = mbase + r;
        float v = acc[r];
        if (SCALE) v *= scale[grow];
        if (BIAS)  v += bias[col];
        if (ACT == 1)      v = fmaxf(v, 0.0f);
        else if (ACT == 2) v = tanhf(v);
        out[(long)grow * ldo + col] = v;
    }
}

// ---------------- edge scatter: agg[dst] += hs[src]  (32 lanes x float4 per edge) ---------------
__global__ void scatter_kernel(const float* __restrict__ hs, const int* __restrict__ src,
                               const int* __restrict__ dst, float* __restrict__ agg, int E) {
    long g = (long)blockIdx.x * blockDim.x + threadIdx.x;
    int e = (int)(g >> 5);
    if (e >= E) return;
    int c = (int)(g & 31) * 4;
    int s = src[e], d = dst[e];
    const float4 v = *(const float4*)(hs + (long)s * FDIM + c);
    float* o = agg + (long)d * FDIM + c;
    atomicAdd(o + 0, v.x);
    atomicAdd(o + 1, v.y);
    atomicAdd(o + 2, v.z);
    atomicAdd(o + 3, v.w);
}

// ---------------- h = relu(agg * cd[row] + b[col]) ----------------
__global__ void relu_norm_bias_kernel(const float* __restrict__ agg, const float* __restrict__ cd,
                                      const float* __restrict__ b, float* __restrict__ out, long total) {
    long t = (long)blockIdx.x * blockDim.x + threadIdx.x;
    if (t >= total) return;
    int n = (int)(t >> 7);     // FDIM = 128
    int f = (int)(t & 127);
    out[t] = fmaxf(agg[t] * cd[n] + b[f], 0.0f);
}

extern "C" void kernel_launch(void* const* d_in, const int* in_sizes, int n_in,
                              void* d_out, int out_size, void* d_ws, size_t ws_size,
                              hipStream_t stream) {
    const float* features = (const float*)d_in[0];
    const float* W0  = (const float*)d_in[1];
    const float* b0  = (const float*)d_in[2];
    const float* W1  = (const float*)d_in[3];
    const float* b1  = (const float*)d_in[4];
    const float* Ws1 = (const float*)d_in[5];
    const float* bs1 = (const float*)d_in[6];
    const float* Ws2 = (const float*)d_in[7];
    const float* bs2 = (const float*)d_in[8];
    const int* src   = (const int*)d_in[9];
    const int* dst   = (const int*)d_in[10];
    const int* users = (const int*)d_in[11];

    float* ws = (float*)d_ws;
    const long NF = (long)N_NODES * FDIM;          // 6,400,000 floats
    float* bufA   = ws;                            // hpre1 / h1 / agg2
    float* bufB   = ws + NF;                       // agg1 / hpre2 / T
    float* degout = ws + 2 * NF;
    float* degin  = degout + N_NODES;
    float* cs     = degin + N_NODES;
    float* cd     = cs + N_NODES;
    bf16_t* pW0   = (bf16_t*)(cd + N_NODES);       // 128x128 bf16 packed
    bf16_t* pW1   = pW0 + 128 * 128;
    bf16_t* pWs1  = pW1 + 128 * 128;               // 128x64
    bf16_t* pWs2  = pWs1 + 128 * 64;               // 64x64

    float* R_out = (float*)d_out;                     // 10000 x 64
    float* h_out = (float*)d_out + N_USERS * OUT_DIM; // 50000 x 128

    // degrees + symmetric norms
    hipMemsetAsync(degout, 0, 2 * N_NODES * sizeof(float), stream);
    degrees_kernel<<<(N_EDGES + 255) / 256, 256, 0, stream>>>(src, dst, degout, degin, N_EDGES);
    norm_kernel<<<(N_NODES + 255) / 256, 256, 0, stream>>>(degout, degin, cs, cd, N_NODES);

    // pack weights into WMMA fragment order (bf16)
    pack_b_kernel<<<(4 * 8 * 512 + 255) / 256, 256, 0, stream>>>(W0,  pW0,  4, 8, 128);
    pack_b_kernel<<<(4 * 8 * 512 + 255) / 256, 256, 0, stream>>>(W1,  pW1,  4, 8, 128);
    pack_b_kernel<<<(4 * 4 * 512 + 255) / 256, 256, 0, stream>>>(Ws1, pWs1, 4, 4, 64);
    pack_b_kernel<<<(2 * 4 * 512 + 255) / 256, 256, 0, stream>>>(Ws2, pWs2, 2, 4, 64);

    const int MT  = N_NODES / 16;   // 3125 (exact)
    const int MTu = N_USERS / 16;   // 625  (exact)
    const int SCB = (int)((long)N_EDGES * 32 / 256); // 200000 blocks
    const int EWB = (int)(NF / 256);                 // 25000 blocks

    // ---- Layer 1: hpre1 = (features @ W0) * cs ; agg ; h1 = relu(agg*cd + b0)
    gemm_wmma_bf16<4, 8, true, false, 0, false>
        <<<MT, 256, 0, stream>>>(features, FDIM, nullptr, pW0, cs, nullptr, bufA, FDIM);
    hipMemsetAsync(bufB, 0, NF * sizeof(float), stream);
    scatter_kernel<<<SCB, 256, 0, stream>>>(bufA, src, dst, bufB, N_EDGES);
    relu_norm_bias_kernel<<<EWB, 256, 0, stream>>>(bufB, cd, b0, bufA, NF);

    // ---- Layer 2: hpre2 = (h1 @ W1) * cs ; agg ; h = relu(agg*cd + b1) -> d_out tail
    gemm_wmma_bf16<4, 8, true, false, 0, false>
        <<<MT, 256, 0, stream>>>(bufA, FDIM, nullptr, pW1, cs, nullptr, bufB, FDIM);
    hipMemsetAsync(bufA, 0, NF * sizeof(float), stream);
    scatter_kernel<<<SCB, 256, 0, stream>>>(bufB, src, dst, bufA, N_EDGES);
    relu_norm_bias_kernel<<<EWB, 256, 0, stream>>>(bufA, cd, b1, h_out, NF);

    // ---- SR head: T = tanh(h[users] @ Ws1 + bs1) ; R = T @ Ws2 + bs2
    gemm_wmma_bf16<4, 4, false, true, 2, true>
        <<<MTu, 128, 0, stream>>>(h_out, FDIM, users, pWs1, nullptr, bs1, bufB, OUT_DIM);
    gemm_wmma_bf16<2, 4, false, true, 0, false>
        <<<MTu, 128, 0, stream>>>(bufB, OUT_DIM, nullptr, pWs2, nullptr, bs2, R_out, OUT_DIM);
}
